// RWGAT_3358664425978
// MI455X (gfx1250) — matrix operand order repaired and checked by reading, not compiled, not run on hardware
//
#include <hip/hip_runtime.h>
#include <cfloat>

#define N_NODES 50000
#define N_EDGES 800000
#define ETOT    (N_EDGES + N_NODES)
#define FDIM    128
#define HEADS   8
#define DIMH    16
#define NEG_SLOPE 0.2f

typedef __attribute__((ext_vector_type(16))) _Float16 v16h;
typedef __attribute__((ext_vector_type(8)))  float    v8f;

// ---- order-preserving float<->uint encoding for atomic max ----
__device__ __forceinline__ unsigned enc_f32(float f) {
  unsigned u = __float_as_uint(f);
  return (u & 0x80000000u) ? ~u : (u | 0x80000000u);
}
__device__ __forceinline__ float dec_f32(unsigned u) {
  return __uint_as_float((u & 0x80000000u) ? (u & 0x7fffffffu) : ~u);
}
__device__ __forceinline__ float lrelu(float x) {
  return x > 0.f ? x : NEG_SLOPE * x;
}

// ============================================================
// h = X (N x 128) @ W (128 x 128); f32 in/out, f16 WMMA inner.
// One block = 16 node rows; 8 waves = 8 head column tiles.
// ============================================================
__global__ void __launch_bounds__(256) gat_gemm_wmma(
    const float* __restrict__ X, const float* __restrict__ W,
    float* __restrict__ H) {
  __shared__ _Float16 sW[FDIM * FDIM];  // transposed: sW[n*128 + k]
  __shared__ _Float16 sX[16 * FDIM];    // sX[m*128 + k]
  const int tid  = threadIdx.x;
  const int row0 = blockIdx.x * 16;

  for (int idx = tid; idx < FDIM * FDIM; idx += 256) {
    int k = idx >> 7, n = idx & 127;
    sW[n * FDIM + k] = (_Float16)W[idx];        // W row-major [k][n]
  }
  for (int idx = tid; idx < 16 * FDIM; idx += 256)
    sX[idx] = (_Float16)X[(size_t)row0 * FDIM + idx];
  __syncthreads();

  const int wave = tid >> 5;         // head / 16-col tile
  const int lane = tid & 31;
  const int half = lane >> 4;        // 0: lanes 0-15, 1: lanes 16-31
  const int lm   = lane & 15;
  const int ncol = wave * 16 + lm;

  v8f c = {};
  #pragma unroll
  for (int kb = 0; kb < FDIM; kb += 32) {
    v16h a, b;
    // A 16x32 f16 layout: lane half selects K sub-blocks {0-7,16-23} / {8-15,24-31}
    #pragma unroll
    for (int t = 0; t < 8; ++t)
      a[t] = sX[lm * FDIM + kb + half * 8 + t];
    #pragma unroll
    for (int t = 8; t < 16; ++t)
      a[t] = sX[lm * FDIM + kb + 8 + half * 8 + t];
    // B 32x16 f16 layout: element t = B[K = kb + half*8 + t + (t >= 8 ? 8 : 0)][N = ncol]
    #pragma unroll
    for (int t = 0; t < 16; ++t)
      b[t] = sW[ncol * FDIM + kb + half * 8 + t + (t >= 8 ? 8 : 0)];
    c = __builtin_amdgcn_wmma_f32_16x16x32_f16(false, a, false, b,
                                               (short)0, c, false, false);
  }
  // C/D layout: VGPR r -> row (half*8 + r), col = lm
  #pragma unroll
  for (int r = 0; r < 8; ++r)
    H[(size_t)(row0 + half * 8 + r) * FDIM + ncol] = c[r];
}

// out := bias broadcast; m := -inf (encoded); denom := 0
__global__ void gat_init(float* __restrict__ out, const float* __restrict__ bias,
                         unsigned* __restrict__ menc, float* __restrict__ denom) {
  int t = blockIdx.x * 256 + threadIdx.x;
  if (t >= N_NODES * FDIM) return;
  int c = t & 127;
  out[t] = bias[c];
  if (c < HEADS) {
    int n = t >> 7;
    menc[n * HEADS + c]  = enc_f32(-FLT_MAX);
    denom[n * HEADS + c] = 0.f;
  }
}

// e_src[n,h] = <h[n,h,:], a_src[h,:]>, same for e_dst
__global__ void gat_edge_logits(const float* __restrict__ H,
                                const float* __restrict__ aS,
                                const float* __restrict__ aD,
                                float* __restrict__ eS, float* __restrict__ eD) {
  int t = blockIdx.x * 256 + threadIdx.x;
  if (t >= N_NODES * HEADS) return;
  int n = t >> 3, h = t & 7;
  const float* hp = H + (size_t)n * FDIM + h * DIMH;
  float s = 0.f, d = 0.f;
  #pragma unroll
  for (int i = 0; i < DIMH; ++i) {
    float v = hp[i];
    s += v * aS[h * DIMH + i];
    d += v * aD[h * DIMH + i];
  }
  eS[t] = s; eD[t] = d;
}

__device__ __forceinline__ void edge_sd(const int* __restrict__ ei, int i,
                                        int& s, int& d) {
  if (i < N_EDGES) { s = ei[i]; d = ei[N_EDGES + i]; }
  else             { s = d = i - N_EDGES; }      // self loops
}

__global__ void gat_edge_max(const int* __restrict__ ei,
                             const float* __restrict__ eS,
                             const float* __restrict__ eD,
                             unsigned* __restrict__ menc) {
  int i = blockIdx.x * 256 + threadIdx.x;
  if (i >= ETOT) return;
  int s, d; edge_sd(ei, i, s, d);
  #pragma unroll
  for (int h = 0; h < HEADS; ++h) {
    float e = lrelu(eS[s * HEADS + h] + eD[d * HEADS + h]);
    atomicMax(&menc[d * HEADS + h], enc_f32(e));
  }
}

__global__ void gat_edge_sum(const int* __restrict__ ei,
                             const float* __restrict__ eS,
                             const float* __restrict__ eD,
                             const unsigned* __restrict__ menc,
                             float* __restrict__ denom) {
  int i = blockIdx.x * 256 + threadIdx.x;
  if (i >= ETOT) return;
  int s, d; edge_sd(ei, i, s, d);
  #pragma unroll
  for (int h = 0; h < HEADS; ++h) {
    float e = lrelu(eS[s * HEADS + h] + eD[d * HEADS + h]);
    float w = expf(e - dec_f32(menc[d * HEADS + h]));
    atomicAdd(&denom[d * HEADS + h], w);
  }
}

// one wave per edge; 4 features per lane; out[dst] += h[src] * alpha
__global__ void gat_edge_scatter(const int* __restrict__ ei,
                                 const float* __restrict__ H,
                                 const float* __restrict__ eS,
                                 const float* __restrict__ eD,
                                 const unsigned* __restrict__ menc,
                                 const float* __restrict__ denom,
                                 float* __restrict__ out) {
  long gid = (long)blockIdx.x * 256 + threadIdx.x;
  int edge = (int)(gid >> 5);
  if (edge >= ETOT) return;
  int lane = threadIdx.x & 31;
  int s, d; edge_sd(ei, edge, s, d);
  int h = lane >> 2;                       // 4 features/lane within one head
  float e     = lrelu(eS[s * HEADS + h] + eD[d * HEADS + h]);
  float w     = expf(e - dec_f32(menc[d * HEADS + h]));
  float alpha = w / (denom[d * HEADS + h] + 1e-16f);
  const float4 hv = *(const float4*)(H + (size_t)s * FDIM + lane * 4);
  float* op = out + (size_t)d * FDIM + lane * 4;
  atomicAdd(op + 0, hv.x * alpha);
  atomicAdd(op + 1, hv.y * alpha);
  atomicAdd(op + 2, hv.z * alpha);
  atomicAdd(op + 3, hv.w * alpha);
}

// one wave per node; in-place log_softmax over 128 features
__global__ void gat_log_softmax(float* __restrict__ out) {
  long gid = (long)blockIdx.x * 256 + threadIdx.x;
  int node = (int)(gid >> 5);
  if (node >= N_NODES) return;
  int lane = threadIdx.x & 31;
  float* p = out + (size_t)node * FDIM + lane * 4;
  float4 v = *(float4*)p;
  float mx = fmaxf(fmaxf(v.x, v.y), fmaxf(v.z, v.w));
  #pragma unroll
  for (int off = 16; off > 0; off >>= 1)
    mx = fmaxf(mx, __shfl_xor(mx, off, 32));
  float sum = expf(v.x - mx) + expf(v.y - mx) + expf(v.z - mx) + expf(v.w - mx);
  #pragma unroll
  for (int off = 16; off > 0; off >>= 1)
    sum += __shfl_xor(sum, off, 32);
  float lse = mx + logf(sum);
  v.x -= lse; v.y -= lse; v.z -= lse; v.w -= lse;
  *(float4*)p = v;
}

static void run_layer(const float* xin, const float* W, const float* aS,
                      const float* aD, const float* bias, const int* ei,
                      float* hbuf, float* eS, float* eD, unsigned* menc,
                      float* denom, float* out, hipStream_t stream) {
  gat_gemm_wmma<<<N_NODES / 16, 256, 0, stream>>>(xin, W, hbuf);
  gat_init<<<(N_NODES * FDIM + 255) / 256, 256, 0, stream>>>(out, bias, menc, denom);
  gat_edge_logits<<<(N_NODES * HEADS + 255) / 256, 256, 0, stream>>>(hbuf, aS, aD, eS, eD);
  gat_edge_max<<<(ETOT + 255) / 256, 256, 0, stream>>>(ei, eS, eD, menc);
  gat_edge_sum<<<(ETOT + 255) / 256, 256, 0, stream>>>(ei, eS, eD, menc, denom);
  gat_edge_scatter<<<((long)ETOT * 32 + 255) / 256, 256, 0, stream>>>(
      ei, hbuf, eS, eD, menc, denom, out);
}

extern "C" void kernel_launch(void* const* d_in, const int* in_sizes, int n_in,
                              void* d_out, int out_size, void* d_ws, size_t ws_size,
                              hipStream_t stream) {
  const float* x   = (const float*)d_in[0];
  const int*   ei  = (const int*)  d_in[1];
  const float* W1  = (const float*)d_in[2];
  const float* aS1 = (const float*)d_in[3];
  const float* aD1 = (const float*)d_in[4];
  const float* b1  = (const float*)d_in[5];
  const float* W2  = (const float*)d_in[6];
  const float* aS2 = (const float*)d_in[7];
  const float* aD2 = (const float*)d_in[8];
  const float* b2  = (const float*)d_in[9];
  float* outp = (float*)d_out;

  float* hbuf = (float*)d_ws;
  float* buf1 = hbuf + (size_t)N_NODES * FDIM;
  float* buf2 = buf1 + (size_t)N_NODES * FDIM;
  float* eS   = buf2 + (size_t)N_NODES * FDIM;
  float* eD   = eS   + (size_t)N_NODES * HEADS;
  unsigned* menc = (unsigned*)(eD + (size_t)N_NODES * HEADS);
  float* denom   = (float*)(menc + (size_t)N_NODES * HEADS);

  run_layer(x,    W1, aS1, aD1, b1, ei, hbuf, eS, eD, menc, denom, buf1, stream);
  run_layer(buf1, W2, aS2, aD2, b2, ei, hbuf, eS, eD, menc, denom, buf2, stream);
  run_layer(buf2, W2, aS2, aD2, b2, ei, hbuf, eS, eD, menc, denom, outp, stream);
  gat_log_softmax<<<N_NODES / 8, 256, 0, stream>>>(outp);
}
